// DPRNN_TAC_17128329576693
// MI455X (gfx1250) — compile-verified, matrix operand order checked by reading, not run on hardware
//
#include <hip/hip_runtime.h>

typedef __attribute__((ext_vector_type(16))) _Float16 v16h;
typedef __attribute__((ext_vector_type(8)))  float    v8f;
typedef __attribute__((ext_vector_type(4)))  unsigned int v4u;
typedef __attribute__((ext_vector_type(8)))  int      v8i;
typedef __attribute__((ext_vector_type(4)))  int      v4i;

static constexpr int kD1 = 100, kD2 = 200, kPQ = kD1 * kD2;
static constexpr int kBC = 8;              // B*CH
static constexpr int kB = 2, kCH = 4;
static constexpr int kE = 384;             // 3*H
static constexpr int kTOT = kBC * 64 * kPQ;  // 10,240,000

#define CDIV(a, b) (((a) + (b) - 1) / (b))

#if __has_builtin(__builtin_amdgcn_tensor_load_to_lds) && \
    __has_builtin(__builtin_amdgcn_s_wait_tensorcnt) &&    \
    __has_builtin(__builtin_amdgcn_groupstaticsize)
#define USE_TDM 1
#endif

__device__ __forceinline__ v8f wmma_f16(v16h a, v16h b, v8f c) {
  return __builtin_amdgcn_wmma_f32_16x16x32_f16(false, a, false, b, (short)0, c,
                                                false, false);
}

// A-fragment (16x32 f16, M x K) from row-major f16 [.., lda] per CDNA5 layout:
// lane 0-15: K 0-7 & 16-23; lane 16-31: K 8-15 & 24-31, M = lane&15.
__device__ __forceinline__ v16h load_a_frag(const _Float16* base, int lda) {
  const int lane = threadIdx.x & 31;
  const int m = lane & 15;
  const int koff = (lane >> 4) << 3;
  const _Float16* p = base + m * lda + koff;
  v16h a;
#pragma unroll
  for (int j = 0; j < 8; ++j) { a[j] = p[j]; a[j + 8] = p[16 + j]; }
  return a;
}

// B-fragment pre-packed: 32 lanes x 16 f16 contiguous per fragment.
__device__ __forceinline__ v16h load_b_frag(const _Float16* frag) {
  const int lane = threadIdx.x & 31;
  return *((const v16h*)frag + lane);
}

__device__ __forceinline__ float preluf(float x, float a) { return x >= 0.f ? x : a * x; }
__device__ __forceinline__ float sigmf(float x) { return 1.f / (1.f + __expf(-x)); }

// ---- pack weights [Nout,K] f32 (row stride ldw) into WMMA B-fragment f16 layout
__global__ void frag_pack_w(const float* __restrict__ W, int ldw, int Nout, int K,
                            _Float16* __restrict__ dst) {
  const int gid = blockIdx.x * blockDim.x + threadIdx.x;
  if (gid >= Nout * K) return;
  const int el = gid & 511, f = gid >> 9;
  const int nkt = K >> 5;
  const int ntile = f / nkt, kt = f % nkt;
  const int lane = el >> 4, j = el & 15;
  const int n = ntile * 16 + (lane & 15);
  const int koff = (lane >> 4) << 3;
  const int kl = (j < 8) ? (koff + j) : (16 + koff + (j - 8));
  dst[gid] = (_Float16)W[(size_t)n * ldw + kt * 32 + kl];
}

// ---- pack the [BC,64,D1,D2] state tensor into f16 GEMM rows
// mode 0: row-LSTM   rows = (bc*D2+q)*D1+p
// mode 1: col-LSTM / final rows = (bc*D1+p)*D2+q   (optional PReLU via alpha)
// mode 2: TAC rows = (((b*D1+p)*D2+q)*4+c)
__global__ void pack_seq(const float* __restrict__ src, _Float16* __restrict__ dst,
                         int mode, const float* __restrict__ alpha) {
  const int gid = blockIdx.x * blockDim.x + threadIdx.x;
  if (gid >= kTOT) return;
  const int n = gid & 63;
  const int r = gid >> 6;
  int bc, p, q;
  if (mode == 0) {
    p = r % kD1; int r2 = r / kD1; q = r2 % kD2; bc = r2 / kD2;
  } else if (mode == 1) {
    q = r % kD2; int r2 = r / kD2; p = r2 % kD1; bc = r2 / kD1;
  } else {
    int c = r & 3; int r2 = r >> 2; q = r2 % kD2;
    int r3 = r2 / kD2; p = r3 % kD1; bc = (r3 / kD1) * 4 + c;
  }
  float v = src[((size_t)(bc * 64 + n) * kD1 + p) * kD2 + q];
  if (alpha) v = preluf(v, *alpha);
  dst[gid] = (_Float16)v;
}

// ---- generic WMMA GEMM: C[M,Nout] = act(A[M,K] * Wfrag^T + bias)
template <typename OutT>
__global__ void wmma_gemm(const _Float16* __restrict__ A, int lda,
                          const _Float16* __restrict__ Bf,
                          const float* __restrict__ bias,
                          const float* __restrict__ alpha, OutT* __restrict__ C,
                          int ldc, int M, int Nout, int K) {
  const int wave = blockIdx.x * (blockDim.x >> 5) + (threadIdx.x >> 5);
  const int ngroups = Nout >> 6;
  const int mtiles = M >> 4;
  if (wave >= mtiles * ngroups) return;
  const int mt = wave / ngroups, ng = wave % ngroups;
  const int lane = threadIdx.x & 31;
  const int nl = lane & 15, koff8 = (lane >> 4) << 3;
  const int nkt = K >> 5;
  v8f acc[4];
#pragma unroll
  for (int nt = 0; nt < 4; ++nt) {
    const float bv = bias ? bias[ng * 64 + nt * 16 + nl] : 0.f;
#pragma unroll
    for (int r = 0; r < 8; ++r) acc[nt][r] = bv;
  }
  const _Float16* Abase = A + (size_t)mt * 16 * lda;
  for (int kt = 0; kt < nkt; ++kt) {
    v16h a = load_a_frag(Abase + kt * 32, lda);
#pragma unroll
    for (int nt = 0; nt < 4; ++nt) {
      const _Float16* bf = Bf + ((size_t)((ng * 4 + nt) * nkt + kt)) * 512;
      acc[nt] = wmma_f16(a, load_b_frag(bf), acc[nt]);
    }
  }
  const float al = alpha ? *alpha : 0.f;
#pragma unroll
  for (int nt = 0; nt < 4; ++nt) {
    const int ncol = ng * 64 + nt * 16 + nl;
#pragma unroll
    for (int r = 0; r < 8; ++r) {
      float v = acc[nt][r];
      if (alpha) v = preluf(v, al);
      C[(size_t)(mt * 16 + r + koff8) * ldc + ncol] = (OutT)v;
    }
  }
}

// ---- persistent LSTM scan: 16 sequences/block, Whh fragments staged into
// dynamic LDS by the Tensor Data Mover, xp = x*Wih^T precomputed. Wave w owns
// gate tiles {w,8+w,16+w,24+w} == i/f/g/o for hidden units j = w*16+nl, so the
// cell update runs entirely in registers.
__global__ void lstm_scan(const float* __restrict__ XP, int T,
                          const _Float16* __restrict__ WhhF,
                          const float* __restrict__ bih,
                          const float* __restrict__ bhh,
                          _Float16* __restrict__ Hout, int dir) {
  extern __shared__ _Float16 wlds[];        // 512*128 f16 fragments (128 KB)
  __shared__ _Float16 hsm[16 * 128];        // h_{t-1} (f16)
  const int tid = threadIdx.x;
  const int s0 = blockIdx.x * 16;

#ifdef USE_TDM
  if (tid < 32) {  // wave 0 drives the TDM; EXEC is ignored by tensor ops
    const unsigned ldsOff = (unsigned)__builtin_amdgcn_groupstaticsize();
    const unsigned long long ga = (unsigned long long)(size_t)WhhF;
    v4u g0;
    g0[0] = 1u;                                        // count=1, user D#
    g0[1] = ldsOff;                                    // lds_addr (dyn base)
    g0[2] = (unsigned)(ga & 0xffffffffu);              // global_addr[31:0]
    g0[3] = (unsigned)((ga >> 32) & 0x01ffffffu) | (2u << 30);  // [56:32]|type=2
    // 1-D tile: data_size=8B, tensor_dim0=tile_dim0=16384 units (=128 KB)
    v8i g1;
    g1[0] = (int)(3u << 16);           // data_size = 3 (8 bytes)
    g1[1] = (int)(16384u << 16);       // tensor_dim0[15:0] << 16
    g1[2] = (int)(1u << 16);           // tensor_dim0[31:16]=0 | tensor_dim1=1
    g1[3] = (int)(16384u << 16);       // tile_dim0 = 16384
    g1[4] = 1;                         // tile_dim1 = 1
    g1[5] = 16384;                     // tensor_dim0_stride[31:0]
    g1[6] = 0;
    g1[7] = 0;
    v4i gz = {0, 0, 0, 0};
#if __clang_major__ >= 23
    v8i gz8 = {0, 0, 0, 0, 0, 0, 0, 0};
    __builtin_amdgcn_tensor_load_to_lds(g0, g1, gz, gz, gz8, 0);
#else
    __builtin_amdgcn_tensor_load_to_lds(g0, g1, gz, gz, 0);
#endif
    __builtin_amdgcn_s_wait_tensorcnt(0);
  }
#else
  {  // fallback: cooperative 16B-chunk copy
    const uint4* src = (const uint4*)WhhF;
    uint4* dst = (uint4*)wlds;
    for (int i = tid; i < (512 * 128 * 2) / 16; i += 256) dst[i] = src[i];
  }
#endif
  for (int i = tid; i < 16 * 128; i += 256) hsm[i] = (_Float16)0.f;
  const int w = tid >> 5, lane = tid & 31;
  const int nl = lane & 15, koff8 = (lane >> 4) << 3;
  const int j = w * 16 + nl;
  const float bi = bih[j] + bhh[j];
  const float bf = bih[128 + j] + bhh[128 + j];
  const float bg = bih[256 + j] + bhh[256 + j];
  const float bo = bih[384 + j] + bhh[384 + j];
  float cst[8];
#pragma unroll
  for (int r = 0; r < 8; ++r) cst[r] = 0.f;
  __syncthreads();  // weights staged + h zeroed, visible to all waves
  for (int st = 0; st < T; ++st) {
    const int t = dir ? (T - 1 - st) : st;
    v16h ah0 = load_a_frag(hsm, 128);
    v16h ah1 = load_a_frag(hsm + 32, 128);
    v16h ah2 = load_a_frag(hsm + 64, 128);
    v16h ah3 = load_a_frag(hsm + 96, 128);
    v8f acc[4];
#pragma unroll
    for (int nt = 0; nt < 4; ++nt)
#pragma unroll
      for (int r = 0; r < 8; ++r)
        acc[nt][r] =
            XP[((size_t)(s0 + r + koff8) * T + t) * 512 + nt * 128 + j];
    __syncthreads();  // all hsm reads done before any wave overwrites it
    // software-pipelined B-fragment stream: load frag i+1, wmma frag i
    v16h bcur = load_b_frag(wlds + ((0 * 8 + w) * 4 + 0) * 512);
#pragma unroll
    for (int it = 0; it < 16; ++it) {
      const int nt = it >> 2, kt = it & 3;
      v16h bnext = bcur;
      if (it < 15) {
        const int nt2 = (it + 1) >> 2, kt2 = (it + 1) & 3;
        bnext = load_b_frag(wlds + ((nt2 * 8 + w) * 4 + kt2) * 512);
      }
      const v16h ah = (kt == 0) ? ah0 : (kt == 1) ? ah1 : (kt == 2) ? ah2 : ah3;
      acc[nt] = wmma_f16(ah, bcur, acc[nt]);
      bcur = bnext;
    }
#pragma unroll
    for (int r = 0; r < 8; ++r) {
      const int m = r + koff8;
      const float c =
          sigmf(acc[1][r] + bf) * cst[r] + sigmf(acc[0][r] + bi) * tanhf(acc[2][r] + bg);
      const float h = sigmf(acc[3][r] + bo) * tanhf(c);
      cst[r] = c;
      const _Float16 hh = (_Float16)h;
      hsm[m * 128 + j] = hh;
      Hout[((size_t)(s0 + m) * T + t) * 256 + dir * 128 + j] = hh;
    }
    __syncthreads();  // hsm updates visible to next step
  }
}

// ---- TAC masked channel mean
__global__ void tac_mean(const _Float16* __restrict__ ch,
                         const int* __restrict__ num_mic,
                         _Float16* __restrict__ cm) {
  const int gid = blockIdx.x * blockDim.x + threadIdx.x;
  const int total = kB * kPQ * kE;
  if (gid >= total) return;
  const int e = gid % kE, g = gid / kE;
  const int b = g / kPQ;
  int eff = num_mic[b];
  if (eff <= 0) eff = kCH;
  float s = 0.f;
  for (int c = 0; c < kCH; ++c)
    if (c < eff) s += (float)ch[((size_t)g * 4 + c) * kE + e];
  cm[gid] = (_Float16)(s / (float)eff);
}

__device__ __forceinline__ float gn_fetch(int mode, int bc, int n, int p, int q,
                                          const float* s1, const float* s2,
                                          const float* cbias, float al) {
  if (mode == 0) return s1[((size_t)(bc * kD2 + q) * kD1 + p) * 64 + n];
  if (mode == 1) return s1[((size_t)(bc * kD1 + p) * kD2 + q) * 64 + n];
  const int b = bc >> 2, c = bc & 3;
  const size_t rg = (size_t)(b * kD1 + p) * kD2 + q;
  return preluf(s1[(rg * 4 + c) * 64 + n] + s2[rg * 64 + n] + cbias[n], al);
}

// ---- fused (PReLU+sum) + GroupNorm(1,C) + residual add; one block per bc sample
__global__ void gn_residual(float* __restrict__ out, const float* __restrict__ s1,
                            const float* __restrict__ s2,
                            const float* __restrict__ cbias,
                            const float* __restrict__ alpha,
                            const float* __restrict__ gw,
                            const float* __restrict__ gb, int mode) {
  __shared__ float r0[1024], r1[1024];
  __shared__ float smu, srs;
  const int bc = blockIdx.x, tid = threadIdx.x;
  const int CNT = 64 * kPQ;
  const float al = (mode == 2) ? *alpha : 0.f;
  float ls = 0.f, lq = 0.f;
  for (int idx = tid; idx < CNT; idx += 1024) {
    const int n = idx / kPQ, rem = idx % kPQ;
    const int p = rem / kD2, q = rem % kD2;
    const float v = gn_fetch(mode, bc, n, p, q, s1, s2, cbias, al);
    ls += v; lq += v * v;
  }
  r0[tid] = ls; r1[tid] = lq;
  __syncthreads();
  for (int s = 512; s > 0; s >>= 1) {
    if (tid < s) { r0[tid] += r0[tid + s]; r1[tid] += r1[tid + s]; }
    __syncthreads();
  }
  if (tid == 0) {
    const float mu = r0[0] / (float)CNT;
    const float var = r1[0] / (float)CNT - mu * mu;
    smu = mu; srs = rsqrtf(var + 1e-8f);
  }
  __syncthreads();
  const float mu = smu, rs = srs;
  for (int idx = tid; idx < CNT; idx += 1024) {
    const int n = idx / kPQ, rem = idx % kPQ;
    const int p = rem / kD2, q = rem % kD2;
    const float v = gn_fetch(mode, bc, n, p, q, s1, s2, cbias, al);
    out[((size_t)(bc * 64 + n) * kD1 + p) * kD2 + q] += (v - mu) * rs * gw[n] + gb[n];
  }
}

__global__ void scatter_final(const float* __restrict__ src, float* __restrict__ dst) {
  const int gid = blockIdx.x * blockDim.x + threadIdx.x;
  if (gid >= kTOT) return;
  const int q = gid % kD2;
  int g2 = gid / kD2;
  const int p = g2 % kD1;
  int g3 = g2 / kD1;
  const int oc = g3 % 64, bc = g3 / 64;
  dst[gid] = src[((size_t)(bc * kD1 + p) * kD2 + q) * 64 + oc];
}

extern "C" void kernel_launch(void* const* d_in, const int* in_sizes, int n_in,
                              void* d_out, int out_size, void* d_ws, size_t ws_size,
                              hipStream_t stream) {
  (void)in_sizes; (void)n_in; (void)out_size; (void)ws_size;
  const float* x        = (const float*)d_in[0];
  const int*   num_mic  = (const int*)d_in[1];
  const float* row_Wih  = (const float*)d_in[2];
  const float* row_Whh  = (const float*)d_in[3];
  const float* row_bih  = (const float*)d_in[4];
  const float* row_bhh  = (const float*)d_in[5];
  const float* row_pw   = (const float*)d_in[6];
  const float* row_pb   = (const float*)d_in[7];
  const float* col_Wih  = (const float*)d_in[8];
  const float* col_Whh  = (const float*)d_in[9];
  const float* col_bih  = (const float*)d_in[10];
  const float* col_bhh  = (const float*)d_in[11];
  const float* col_pw   = (const float*)d_in[12];
  const float* col_pb   = (const float*)d_in[13];
  const float* tr_w = (const float*)d_in[14];
  const float* tr_b = (const float*)d_in[15];
  const float* tr_a = (const float*)d_in[16];
  const float* av_w = (const float*)d_in[17];
  const float* av_b = (const float*)d_in[18];
  const float* av_a = (const float*)d_in[19];
  const float* cc_w = (const float*)d_in[20];
  const float* cc_b = (const float*)d_in[21];
  const float* cc_a = (const float*)d_in[22];
  const float* rn_w = (const float*)d_in[23];
  const float* rn_b = (const float*)d_in[24];
  const float* cn_w = (const float*)d_in[25];
  const float* cn_b = (const float*)d_in[26];
  const float* chn_w = (const float*)d_in[27];
  const float* chn_b = (const float*)d_in[28];
  const float* out_a = (const float*)d_in[29];
  const float* out_w = (const float*)d_in[30];
  const float* out_b = (const float*)d_in[31];

  char* ws = (char*)d_ws;
  size_t off = 0;
  auto alloc = [&](size_t bytes) {
    void* p = ws + off;
    off = (off + bytes + 255) & ~(size_t)255;
    return p;
  };
  // persistent
  float*    BUF   = (float*)alloc((size_t)kTOT * 4);      // state [BC,64,D1,D2]
  _Float16* PACKA = (_Float16*)alloc((size_t)kTOT * 2);   // f16 GEMM rows
  _Float16* WFR   = (_Float16*)alloc((size_t)524288 * 2); // weight fragments
  // phase union: LSTM {XP,SEQH,SEQP} / TAC {CH16,CM16,CMO,TMP1,TMP2}
  char* U = (char*)alloc((size_t)160000 * 512 * 4 + (size_t)160000 * 256 * 2 +
                         (size_t)160000 * 64 * 4);
  float*    XP   = (float*)U;                                         // 327.68 MB
  _Float16* SEQH = (_Float16*)(U + (size_t)160000 * 512 * 4);         // 81.92 MB
  float*    SEQP = (float*)(U + (size_t)160000 * 512 * 4 + (size_t)160000 * 256 * 2);
  _Float16* CH16 = (_Float16*)U;                                      // 122.88 MB
  _Float16* CM16 = (_Float16*)(U + (size_t)160000 * kE * 2);
  _Float16* CMO  = (_Float16*)(U + (size_t)200000 * kE * 2);
  float*    TMP1 = (float*)(U + (size_t)240000 * kE * 2);
  float*    TMP2 = (float*)(U + (size_t)240000 * kE * 2 + (size_t)160000 * 64 * 4);

  _Float16* WF_WIH  = WFR;            // 512*64
  _Float16* WF_WHH  = WFR + 32768;    // 512*128
  _Float16* WF_PROJ = WFR + 98304;    // 64*256
  _Float16* WF_TR   = WFR + 114688;   // 384*64
  _Float16* WF_AV   = WFR + 139264;   // 384*384
  _Float16* WF_CC1  = WFR + 286720;   // 64*384
  _Float16* WF_CC2  = WFR + 311296;   // 64*384
  _Float16* WF_OUT  = WFR + 335872;   // 64*64

  hipMemcpyAsync(BUF, x, (size_t)kTOT * 4, hipMemcpyDeviceToDevice, stream);

  const size_t kWhhLds = (size_t)512 * 128 * 2;  // 128 KB dynamic LDS

  for (int i = 0; i < 2; ++i) {
    // ---------- intra-segment (row) BiLSTM: 1600 seqs x T=100 ----------
    pack_seq<<<CDIV(kTOT, 256), 256, 0, stream>>>(BUF, PACKA, 0, nullptr);
    for (int d = 0; d < 2; ++d) {
      frag_pack_w<<<CDIV(512 * 64, 256), 256, 0, stream>>>(
          row_Wih + (size_t)(i * 2 + d) * 512 * 64, 64, 512, 64, WF_WIH);
      frag_pack_w<<<CDIV(512 * 128, 256), 256, 0, stream>>>(
          row_Whh + (size_t)(i * 2 + d) * 512 * 128, 128, 512, 128, WF_WHH);
      wmma_gemm<float><<<CDIV(10000 * 8, 8), 256, 0, stream>>>(
          PACKA, 64, WF_WIH, nullptr, nullptr, XP, 512, 160000, 512, 64);
      lstm_scan<<<100, 256, kWhhLds, stream>>>(
          XP, 100, WF_WHH, row_bih + (size_t)(i * 2 + d) * 512,
          row_bhh + (size_t)(i * 2 + d) * 512, SEQH, d);
    }
    frag_pack_w<<<CDIV(64 * 256, 256), 256, 0, stream>>>(
        row_pw + (size_t)i * 64 * 256, 256, 64, 256, WF_PROJ);
    wmma_gemm<float><<<CDIV(10000, 8), 256, 0, stream>>>(
        SEQH, 256, WF_PROJ, row_pb + i * 64, nullptr, SEQP, 64, 160000, 64, 256);
    gn_residual<<<8, 1024, 0, stream>>>(BUF, SEQP, nullptr, nullptr, nullptr,
                                        rn_w + i * 64, rn_b + i * 64, 0);

    // ---------- inter-segment (col) BiLSTM: 800 seqs x T=200 ----------
    pack_seq<<<CDIV(kTOT, 256), 256, 0, stream>>>(BUF, PACKA, 1, nullptr);
    for (int d = 0; d < 2; ++d) {
      frag_pack_w<<<CDIV(512 * 64, 256), 256, 0, stream>>>(
          col_Wih + (size_t)(i * 2 + d) * 512 * 64, 64, 512, 64, WF_WIH);
      frag_pack_w<<<CDIV(512 * 128, 256), 256, 0, stream>>>(
          col_Whh + (size_t)(i * 2 + d) * 512 * 128, 128, 512, 128, WF_WHH);
      wmma_gemm<float><<<CDIV(10000 * 8, 8), 256, 0, stream>>>(
          PACKA, 64, WF_WIH, nullptr, nullptr, XP, 512, 160000, 512, 64);
      lstm_scan<<<50, 256, kWhhLds, stream>>>(
          XP, 200, WF_WHH, col_bih + (size_t)(i * 2 + d) * 512,
          col_bhh + (size_t)(i * 2 + d) * 512, SEQH, d);
    }
    frag_pack_w<<<CDIV(64 * 256, 256), 256, 0, stream>>>(
        col_pw + (size_t)i * 64 * 256, 256, 64, 256, WF_PROJ);
    wmma_gemm<float><<<CDIV(10000, 8), 256, 0, stream>>>(
        SEQH, 256, WF_PROJ, col_pb + i * 64, nullptr, SEQP, 64, 160000, 64, 256);
    gn_residual<<<8, 1024, 0, stream>>>(BUF, SEQP, nullptr, nullptr, nullptr,
                                        cn_w + i * 64, cn_b + i * 64, 1);

    // ---------- TAC ----------
    pack_seq<<<CDIV(kTOT, 256), 256, 0, stream>>>(BUF, PACKA, 2, nullptr);
    frag_pack_w<<<CDIV(kE * 64, 256), 256, 0, stream>>>(
        tr_w + (size_t)i * kE * 64, 64, kE, 64, WF_TR);
    wmma_gemm<_Float16><<<CDIV(10000 * 6, 8), 256, 0, stream>>>(
        PACKA, 64, WF_TR, tr_b + i * kE, tr_a + i, CH16, kE, 160000, kE, 64);
    tac_mean<<<CDIV(40000 * kE, 256), 256, 0, stream>>>(CH16, num_mic, CM16);
    frag_pack_w<<<CDIV(kE * kE, 256), 256, 0, stream>>>(
        av_w + (size_t)i * kE * kE, kE, kE, kE, WF_AV);
    wmma_gemm<_Float16><<<CDIV(2500 * 6, 8), 256, 0, stream>>>(
        CM16, kE, WF_AV, av_b + i * kE, av_a + i, CMO, kE, 40000, kE, kE);
    frag_pack_w<<<CDIV(64 * kE, 256), 256, 0, stream>>>(
        cc_w + (size_t)i * 64 * 768, 768, 64, kE, WF_CC1);
    frag_pack_w<<<CDIV(64 * kE, 256), 256, 0, stream>>>(
        cc_w + (size_t)i * 64 * 768 + kE, 768, 64, kE, WF_CC2);
    wmma_gemm<float><<<CDIV(10000, 8), 256, 0, stream>>>(
        CH16, kE, WF_CC1, nullptr, nullptr, TMP1, 64, 160000, 64, kE);
    wmma_gemm<float><<<CDIV(2500, 8), 256, 0, stream>>>(
        CMO, kE, WF_CC2, nullptr, nullptr, TMP2, 64, 40000, 64, kE);
    gn_residual<<<8, 1024, 0, stream>>>(BUF, TMP1, TMP2, cc_b + i * 64, cc_a + i,
                                        chn_w + i * 64, chn_b + i * 64, 2);
  }

  // ---------- final PReLU + 1x1 conv ----------
  pack_seq<<<CDIV(kTOT, 256), 256, 0, stream>>>(BUF, PACKA, 1, out_a);
  frag_pack_w<<<CDIV(64 * 64, 256), 256, 0, stream>>>(out_w, 64, 64, 64, WF_OUT);
  wmma_gemm<float><<<CDIV(10000, 8), 256, 0, stream>>>(
      PACKA, 64, WF_OUT, out_b, nullptr, TMP1, 64, 160000, 64, 64);
  scatter_final<<<CDIV(kTOT, 256), 256, 0, stream>>>(TMP1, (float*)d_out);
}
